// GMPool_4_67181878444699
// MI455X (gfx1250) — matrix operand order repaired and checked by reading, not compiled, not run on hardware
//
#include <hip/hip_runtime.h>
#include <hip/hip_bf16.h>
#include <math.h>

// Problem constants (from reference)
#define B_    8
#define T_    176
#define R_    116
#define C_    64
#define H_    200
#define HPAD  208      // H padded to 13 * 16
#define NTILES 13
#define MAXG_ 176
#define NPAIR 6670     // 116*115/2
#define TAU_  1e-4f
#define TM_   4        // t-rows per WMMA block

typedef __attribute__((ext_vector_type(16))) _Float16 v16h;
typedef __attribute__((ext_vector_type(8)))  float    v8f;

// ---------------------------------------------------------------------------
// Prep: pack fc1_w into the per-lane WMMA B-operand order used here (f16), pad
// fc2_w / fc1_b to HPAD. B element j of lane L: K = j + 16*(L>=16), N = L&15.
// Buffer index: ((nt*2 + kt)*32 + lane)*16 + j, channel k = kt*32 + K.
// ---------------------------------------------------------------------------
__global__ void k_prep(const float* __restrict__ fc1_w,
                       const float* __restrict__ fc1_b,
                       const float* __restrict__ fc2_w,
                       _Float16* __restrict__ fc1B,
                       float* __restrict__ fc2pad,
                       float* __restrict__ fc1bpad) {
  int tid = blockIdx.x * blockDim.x + threadIdx.x;
  int nthreads = gridDim.x * blockDim.x;
  const int total = NTILES * 2 * 32 * 16;
  for (int idx = tid; idx < total; idx += nthreads) {
    int nt   = idx >> 10;          // / (2*32*16)
    int rem  = idx & 1023;
    int kt   = rem >> 9;
    int lane = (rem >> 4) & 31;
    int j    = rem & 15;
    int n    = nt * 16 + (lane & 15);
    int k    = kt * 32 + j + 16 * (lane >> 4);
    float v  = (n < H_) ? fc1_w[n * C_ + k] : 0.0f;
    fc1B[idx] = (_Float16)v;
  }
  for (int i = tid; i < HPAD; i += nthreads) {
    fc2pad[i]  = (i < H_) ? fc2_w[i] : 0.0f;
    fc1bpad[i] = (i < H_) ? fc1_b[i] : 0.0f;
  }
}

// ---------------------------------------------------------------------------
// out_cnn[b,t,c] = X[b,t,:] . conv_w[c,:] + conv_b[c]   (tiny GEMM, VALU)
// ---------------------------------------------------------------------------
__global__ void k_outcnn(const float* __restrict__ X,
                         const float* __restrict__ conv_w,
                         const float* __restrict__ conv_b,
                         float* __restrict__ out_cnn) {
  __shared__ float xrow[R_];
  int b = blockIdx.y, t = blockIdx.x;
  for (int r = threadIdx.x; r < R_; r += blockDim.x)
    xrow[r] = X[((size_t)b * T_ + t) * R_ + r];
  __syncthreads();
  int c = threadIdx.x;
  if (c < C_) {
    float acc = conv_b[c];
    for (int r = 0; r < R_; ++r) acc = fmaf(xrow[r], conv_w[c * R_ + r], acc);
    out_cnn[((size_t)b * T_ + t) * C_ + c] = acc;
  }
}

// ---------------------------------------------------------------------------
// Fused dist -> fc1(WMMA f16, f32 acc) -> relu -> fc2 dot -> sigmoid -> gm.
// One wave32 per (b, 4-wide t tile, 16-wide s tile). K=64 = 2 WMMA ktiles.
// A layout (16-bit 16x32): element j of lane L holds
//   M = L&15, K = j + 16*(L>=16) (contiguous k, same as fc1B).
// C/D layout: element e of lane L holds M = e + 8*(L>=16), N = L&15.
// N-tile loop is NOT unrolled (unroll 1): exactly one live v8f accumulator,
// folded into the fc2 dot immediately -> no scratch spills.
// ---------------------------------------------------------------------------
__global__ __launch_bounds__(32) void k_clf_wmma(
    const float* __restrict__ out_cnn,
    const _Float16* __restrict__ fc1B,
    const float* __restrict__ fc1bpad,
    const float* __restrict__ fc2pad,
    const float* __restrict__ fc2_b,
    float* __restrict__ gm) {
  __shared__ float sT[TM_ * C_];
  __shared__ float sS[16 * C_];
  int lane = threadIdx.x;
  int s0 = blockIdx.x * 16;
  int t0 = blockIdx.y * TM_;
  int b  = blockIdx.z;
  const float* base = out_cnn + (size_t)b * T_ * C_;
  for (int i = lane; i < TM_ * C_; i += 32)
    sT[i] = base[(t0 + (i >> 6)) * C_ + (i & 63)];
  for (int i = lane; i < 16 * C_; i += 32)
    sS[i] = base[(s0 + (i >> 6)) * C_ + (i & 63)];
  __syncthreads();

  int M  = lane & 15;      // A-row (= s within tile), also C-col N
  int hi = lane >> 4;
  v16h a0[TM_], a1[TM_];
#pragma unroll
  for (int tm = 0; tm < TM_; ++tm) {
#pragma unroll
    for (int j = 0; j < 16; ++j) {
      int K = j + (hi << 4);
      float d0 = sT[tm * C_ + K] - sS[M * C_ + K];
      a0[tm][j] = (_Float16)sqrtf(d0 * d0 + 1e-9f);
      int c1 = K + 32;
      float d1 = sT[tm * C_ + c1] - sS[M * C_ + c1];
      a1[tm][j] = (_Float16)sqrtf(d1 * d1 + 1e-9f);
    }
  }

  const v16h* Bp = (const v16h*)fc1B;
  float cl[TM_][8];
#pragma unroll
  for (int tm = 0; tm < TM_; ++tm)
#pragma unroll
    for (int e = 0; e < 8; ++e) cl[tm][e] = 0.0f;

#pragma unroll 1
  for (int nt = 0; nt < NTILES; ++nt) {
    v16h b0 = Bp[(nt * 2 + 0) * 32 + lane];
    v16h b1 = Bp[(nt * 2 + 1) * 32 + lane];
    float bias = fc1bpad[nt * 16 + M];
    float w2   = fc2pad[nt * 16 + M];
#pragma unroll
    for (int tm = 0; tm < TM_; ++tm) {
      v8f acc = {};
      acc = __builtin_amdgcn_wmma_f32_16x16x32_f16(false, a0[tm], false, b0,
                                                   (short)0, acc, false, false);
      acc = __builtin_amdgcn_wmma_f32_16x16x32_f16(false, a1[tm], false, b1,
                                                   (short)0, acc, false, false);
#pragma unroll
      for (int e = 0; e < 8; ++e) {
        float h = acc[e] + bias;
        h = h > 0.0f ? h : 0.0f;
        cl[tm][e] = fmaf(h, w2, cl[tm][e]);
      }
    }
  }
  // Reduce over the N dimension (16 lanes of each half-wave)
  for (int off = 1; off < 16; off <<= 1)
#pragma unroll
    for (int tm = 0; tm < TM_; ++tm)
#pragma unroll
      for (int e = 0; e < 8; ++e) cl[tm][e] += __shfl_xor(cl[tm][e], off, 32);

  if ((lane & 15) == 0) {
    float b2 = fc2_b[0];
#pragma unroll
    for (int tm = 0; tm < TM_; ++tm) {
      int t = t0 + tm;
#pragma unroll
      for (int e = 0; e < 8; ++e) {
        int s = s0 + e + (hi << 3);
        float sig = 1.0f / (1.0f + expf(-(cl[tm][e] + b2)));
        gm[((size_t)b * T_ + t) * T_ + s] = (s == t) ? 1.0f : sig;
      }
    }
  }
}

// ---------------------------------------------------------------------------
// Batched symmetric eigensolver: block-parallel cyclic Jacobi, one workgroup
// per batch. A and V both resident in the 320KB CDNA5 LDS (248KB dynamic).
// Then reproduce the reference quirk: vecs stay in ASCENDING eigenvalue
// order while vals are reversed -> S[:,k] = vec_asc[k] * sqrt(val_desc[k]).
// Softmax((S+gumbel)/TAU) -> soft index.
// ---------------------------------------------------------------------------
__global__ __launch_bounds__(256) void k_eigen(const float* __restrict__ gm,
                                               const float* __restrict__ gumbel,
                                               float* __restrict__ sidx) {
  extern __shared__ float sm[];
  float* A = sm;               // T_*T_
  float* V = sm + T_ * T_;     // T_*T_
  __shared__ float rc[88], rs[88];
  __shared__ int   rp[88], rq[88];
  __shared__ float lam[T_], lsorted[T_], scalev[T_];
  __shared__ int   order[T_];

  int b = blockIdx.x;
  int tid = threadIdx.x;
  const float* Gm = gm + (size_t)b * T_ * T_;
  for (int i = tid; i < T_ * T_; i += 256) {
    A[i] = Gm[i];
    V[i] = ((i / T_) == (i % T_)) ? 1.0f : 0.0f;
  }
  __syncthreads();

  const int NSWEEP = 12;
  for (int sweep = 0; sweep < NSWEEP; ++sweep) {
    for (int r = 0; r < T_ - 1; ++r) {
      if (tid < 88) {
        int k  = tid;
        int s2 = T_ - 1 - k;
        int p1 = (k == 0) ? 0 : ((k - 1 + r) % (T_ - 1)) + 1;
        int p2 = ((s2 - 1 + r) % (T_ - 1)) + 1;   // s2 >= 88 > 0
        int p = p1 < p2 ? p1 : p2;
        int q = p1 < p2 ? p2 : p1;
        float app = A[p * T_ + p], aqq = A[q * T_ + q], apq = A[p * T_ + q];
        float c = 1.0f, s = 0.0f;
        if (fabsf(apq) > 1e-13f) {
          float th = (aqq - app) / (2.0f * apq);
          float tt = 1.0f / (fabsf(th) + sqrtf(th * th + 1.0f));
          if (th < 0.0f) tt = -tt;
          c = rsqrtf(tt * tt + 1.0f);
          s = tt * c;
        }
        rc[k] = c; rs[k] = s; rp[k] = p; rq[k] = q;
      }
      __syncthreads();
      // Column rotations on A and V (disjoint column pairs)
      for (int w = tid; w < 88 * T_; w += 256) {
        int k = w / T_;
        int i = w - k * T_;
        int p = rp[k], q = rq[k];
        float c = rc[k], s = rs[k];
        float x = A[i * T_ + p], y = A[i * T_ + q];
        A[i * T_ + p] = c * x - s * y;
        A[i * T_ + q] = s * x + c * y;
        x = V[i * T_ + p]; y = V[i * T_ + q];
        V[i * T_ + p] = c * x - s * y;
        V[i * T_ + q] = s * x + c * y;
      }
      __syncthreads();
      // Row rotations on A (disjoint row pairs)
      for (int w = tid; w < 88 * T_; w += 256) {
        int k = w / T_;
        int j = w - k * T_;
        int p = rp[k], q = rq[k];
        float c = rc[k], s = rs[k];
        float x = A[p * T_ + j], y = A[q * T_ + j];
        A[p * T_ + j] = c * x - s * y;
        A[q * T_ + j] = s * x + c * y;
      }
      __syncthreads();
    }
  }

  if (tid < T_) lam[tid] = A[tid * T_ + tid];
  __syncthreads();
  if (tid < T_) {
    float l = lam[tid];
    int rank = 0;
    for (int j = 0; j < T_; ++j) {
      float lj = lam[j];
      if (lj < l || (lj == l && j < tid)) rank++;
    }
    order[rank] = tid;
    lsorted[rank] = l;
  }
  __syncthreads();
  if (tid < T_) {
    float vd = lsorted[T_ - 1 - tid];        // descending values
    if (vd <= 0.0f) vd = 1e-7f;
    scalev[tid] = sqrtf(vd) * 1000.0f;       // paired with ASCENDING vecs
  }
  __syncthreads();
  if (tid < T_) {
    int i = tid;
    const float* gb = gumbel + ((size_t)b * T_ + i) * T_;
    float mx = -3.402823466e+38f;
    for (int k = 0; k < T_; ++k) {
      float z = (V[i * T_ + order[k]] * scalev[k] + gb[k]) * (1.0f / TAU_);
      mx = fmaxf(mx, z);
    }
    float se = 0.0f, swi = 0.0f;
    for (int k = 0; k < T_; ++k) {
      float z = (V[i * T_ + order[k]] * scalev[k] + gb[k]) * (1.0f / TAU_);
      float e = expf(z - mx);
      se += e;
      swi += e * (float)k;
    }
    sidx[(size_t)b * T_ + i] = swi / se;
  }
}

// ---------------------------------------------------------------------------
// Contiguous-run grouping of soft indices (exact float compares, like ref).
// ---------------------------------------------------------------------------
__global__ void k_groups(const float* __restrict__ sidx,
                         int* __restrict__ gstart, int* __restrict__ glen,
                         int* __restrict__ ngroups) {
  int b = blockIdx.x;
  if (threadIdx.x != 0) return;
  const float* id = sidx + (size_t)b * T_;
  int ng = 0, start = 0;
  float prev = id[0];
  for (int t = 1; t < T_; ++t) {
    float v = id[t];
    if (v != prev) {
      gstart[b * MAXG_ + ng] = start;
      glen[b * MAXG_ + ng]   = t - start;
      ng++; start = t;
    }
    prev = v;
  }
  gstart[b * MAXG_ + ng] = start;
  glen[b * MAXG_ + ng]   = T_ - start;
  ng++;
  ngroups[b] = ng;
}

// ---------------------------------------------------------------------------
// Per (batch, group): cov = Gram(group cols) - P P^T / T  (mean over full T),
// corr = clip(cov / sqrt(d_r d_s)), upper-triangle flat out. Invalid -> 0.
// ---------------------------------------------------------------------------
__global__ __launch_bounds__(256) void k_corr(const float* __restrict__ X,
                                              const int* __restrict__ gstart,
                                              const int* __restrict__ glen,
                                              const int* __restrict__ ngroups,
                                              float* __restrict__ padded) {
  int g = blockIdx.x, b = blockIdx.y;
  int tid = threadIdx.x;
  float* out = padded + ((size_t)b * MAXG_ + g) * NPAIR;
  if (g >= ngroups[b]) {
    for (int i = tid; i < NPAIR; i += 256) out[i] = 0.0f;
    return;
  }
  __shared__ float P[R_], Dg[R_];
  int a = gstart[b * MAXG_ + g];
  int m = glen[b * MAXG_ + g];
  const float* Xb = X + (size_t)b * T_ * R_;
  if (tid < R_) {
    float p = 0.0f, ss = 0.0f;
    for (int i = 0; i < m; ++i) {
      float x = Xb[(size_t)(a + i) * R_ + tid];
      p += x;
      ss = fmaf(x, x, ss);
    }
    P[tid]  = p;
    Dg[tid] = ss - p * p * (1.0f / T_);
  }
  __syncthreads();
  for (int pi = tid; pi < NPAIR; pi += 256) {
    int r = 0, accum = 0;
    while (pi >= accum + (R_ - 1 - r)) { accum += R_ - 1 - r; r++; }
    int s = r + 1 + (pi - accum);
    float cov = 0.0f;
    for (int i = 0; i < m; ++i) {
      const float* row = Xb + (size_t)(a + i) * R_;
      cov = fmaf(row[r], row[s], cov);
    }
    cov -= P[r] * P[s] * (1.0f / T_);
    float prod = fmaxf(Dg[r] * Dg[s], 0.0f);
    float den = sqrtf(prod);
    den = den > 0.0f ? den : 1.0f;
    float corr = cov / den;
    out[pi] = fminf(1.0f, fmaxf(-1.0f, corr));
  }
}

// ---------------------------------------------------------------------------
extern "C" void kernel_launch(void* const* d_in, const int* in_sizes, int n_in,
                              void* d_out, int out_size, void* d_ws, size_t ws_size,
                              hipStream_t stream) {
  const float* X      = (const float*)d_in[0];
  const float* conv_w = (const float*)d_in[1];
  const float* conv_b = (const float*)d_in[2];
  const float* fc1_w  = (const float*)d_in[3];
  const float* fc1_b  = (const float*)d_in[4];
  const float* fc2_w  = (const float*)d_in[5];
  const float* fc2_b  = (const float*)d_in[6];
  const float* gumbel = (const float*)d_in[7];

  // Outputs concatenated: padded (B,176,6670) | grouping_M (B,T,T) | sidx (B,T)
  float* padded = (float*)d_out;
  float* gm     = padded + (size_t)B_ * MAXG_ * NPAIR;
  float* sidx   = gm + (size_t)B_ * T_ * T_;

  // Workspace layout (~400 KB total)
  char* ws = (char*)d_ws;
  float*    out_cnn = (float*)ws;                         // 90112 f32
  _Float16* fc1B    = (_Float16*)(ws + 360448);           // 13312 f16 (32B-aligned)
  float*    fc2pad  = (float*)(ws + 360448 + 26624);      // 208 f32
  float*    fc1bpad = fc2pad + HPAD;                      // 208 f32
  int*      gstart  = (int*)(fc1bpad + HPAD);             // B*176 int
  int*      glen    = gstart + B_ * MAXG_;                // B*176 int
  int*      ngroups = glen + B_ * MAXG_;                  // B int

  k_prep<<<4, 256, 0, stream>>>(fc1_w, fc1_b, fc2_w, fc1B, fc2pad, fc1bpad);
  k_outcnn<<<dim3(T_, B_), 64, 0, stream>>>(X, conv_w, conv_b, out_cnn);
  k_clf_wmma<<<dim3(T_ / 16, T_ / TM_, B_), 32, 0, stream>>>(
      out_cnn, fc1B, fc1bpad, fc2pad, fc2_b, gm);
  (void)hipFuncSetAttribute((const void*)k_eigen,
                            hipFuncAttributeMaxDynamicSharedMemorySize,
                            2 * T_ * T_ * (int)sizeof(float));
  k_eigen<<<B_, 256, 2 * T_ * T_ * sizeof(float), stream>>>(gm, gumbel, sidx);
  k_groups<<<B_, 32, 0, stream>>>(sidx, gstart, glen, ngroups);
  k_corr<<<dim3(MAXG_, B_), 256, 0, stream>>>(X, gstart, glen, ngroups, padded);
}
